// EnergyDistributionCNN_64218351010357
// MI455X (gfx1250) — compile-verified, matrix operand order checked by reading, not run on hardware
//
#include <hip/hip_runtime.h>
#include <hip/hip_bf16.h>

// CDNA5 (gfx1250) wave32 WMMA implementation of the per-pixel 54->32->13 MLP.
// D = A x B + C with v_wmma_f32_16x16x32_f16:
//   M = 16 pixels per wave, layer1 K = 64 (54 padded), N = 32 (2 tiles of 16),
//   layer2 K = 32, N = 16 (13 padded).
// Rotation permutation of the 3x3 taps is applied arithmetically
// ((dy,dx) = R(dir)*(a,b)) so the gather needs no table lookups.

typedef __attribute__((ext_vector_type(16))) _Float16 v16h;
typedef __attribute__((ext_vector_type(8)))  _Float16 v8h;
typedef __attribute__((ext_vector_type(8)))  float    v8f;

#define W_DIM  1024
#define W_MASK 1023
#define N_PIX  (W_DIM * W_DIM)

// 16-bit A-matrix 16x32 layout (ISA 7.12.2): element e of the v16h in lane L
// holds K = e + (e>=8 ? 8 : 0) + (L>=16 ? 8 : 0).  B mirrors it with lane = N.
__device__ __forceinline__ constexpr int kbase(int e) {
    return e + ((e & 8) ? 8 : 0);
}

// ---------------------------------------------------------------------------
// Setup kernel (1 wave): build the layer-1/layer-2 weight B fragments once.
// ws layout: 5 fragments x 32 lanes x v16h (32B)  = 5120 bytes.
//   frag 0: w1 tile n=0..15,  K=0..31      frag 1: w1 tile n=0..15,  K=32..63
//   frag 2: w1 tile n=16..31, K=0..31      frag 3: w1 tile n=16..31, K=32..63
//   frag 4: w2 (n = 0..12 valid),          K=0..31
// ---------------------------------------------------------------------------
__global__ void __launch_bounds__(32)
build_weight_frags(const float* __restrict__ w1,
                   const float* __restrict__ w2,
                   _Float16* __restrict__ wsf)
{
    const int lane = threadIdx.x & 31;
    const int hi   = lane >> 4;
    const int lm   = lane & 15;

    v16h f0, f1, f2, f3, f4;
#pragma unroll
    for (int e = 0; e < 16; ++e) {
        const int k0 = kbase(e) + 8 * hi;       // 0..31
        const int k1 = k0 + 32;                 // 32..63
        f0[e] = (_Float16)w1[lm * 54 + k0];
        f1[e] = (k1 < 54) ? (_Float16)w1[lm * 54 + k1] : (_Float16)0.0f;
        f2[e] = (_Float16)w1[(lm + 16) * 54 + k0];
        f3[e] = (k1 < 54) ? (_Float16)w1[(lm + 16) * 54 + k1] : (_Float16)0.0f;
        f4[e] = (lm < 13) ? (_Float16)w2[lm * 32 + k0] : (_Float16)0.0f;
    }
    v16h* o = (v16h*)wsf;
    o[0 * 32 + lane] = f0;
    o[1 * 32 + lane] = f1;
    o[2 * 32 + lane] = f2;
    o[3 * 32 + lane] = f3;
    o[4 * 32 + lane] = f4;
}

// ---------------------------------------------------------------------------
// Gather the rotated 3x3x6 patch into two A fragments.  KOFF = 8*hi is a
// template constant so ch = k/9 and tap j = k%9 fold at compile time; the
// per-tap element offsets voff[] were precomputed with the rotation basis.
// ---------------------------------------------------------------------------
template<int KOFF>
__device__ __forceinline__ void gather_frags(const float* __restrict__ se,
                                             const float* __restrict__ te,
                                             const float* __restrict__ sr,
                                             const float* __restrict__ hc,
                                             const int* voff, v16h& a0, v16h& a1)
{
#pragma unroll
    for (int e = 0; e < 16; ++e) {
        const int k0 = kbase(e) + KOFF;         // 0..31, always valid
        {
            const int ch = k0 / 9, j = k0 % 9;
            const float* base = (ch == 0) ? se : (ch == 1) ? te :
                                (ch == 2) ? sr : (hc + (ch - 3) * N_PIX);
            a0[e] = (_Float16)base[voff[j]];
        }
        const int k1 = k0 + 32;                 // 32..63, valid while < 54
        if (k1 < 54) {
            const int ch = k1 / 9, j = k1 % 9;
            const float* base = (ch == 0) ? se : (ch == 1) ? te :
                                (ch == 2) ? sr : (hc + (ch - 3) * N_PIX);
            a1[e] = (_Float16)base[voff[j]];
        } else {
            a1[e] = (_Float16)0.0f;
        }
    }
}

__global__ void __launch_bounds__(256)
energy_cnn_wmma_kernel(const float* __restrict__ se,
                       const float* __restrict__ te,
                       const float* __restrict__ sr,
                       const float* __restrict__ hc,
                       const float* __restrict__ rotm,
                       const float* __restrict__ b1,
                       const float* __restrict__ b2,
                       const _Float16* __restrict__ wsf,
                       float* __restrict__ out)
{
    __shared__ _Float16 ldsX[8][16][32];  // layer-1 activations per wave (f16)
    __shared__ float    ldsY[8][16][16];  // layer-2 outputs per wave (f32)

    const int tid  = threadIdx.x;
    const int wv   = tid >> 5;         // wave in block (0..7)
    const int lane = tid & 31;
    const int hi   = lane >> 4;        // lane half (K-offset selector)
    const int lm   = lane & 15;        // M index (A) or N index (B/C)
    const int gw   = blockIdx.x * 8 + wv;
    const int p0   = gw * 16;          // first pixel of this wave's tile

    // Lanes L and L+16 both gather for pixel p0+lm (different K halves).
    const int p  = p0 + lm;
    const int py = p >> 10;
    const int px = p & W_MASK;

    const float TWO_PI = 6.2831853071795864f;
    float ang = fmodf(rotm[p], TWO_PI);
    const int dir = ((int)(ang * (4.0f / TWO_PI))) & 3;

    // rotation basis: ROT[dir] maps tap (a,b) -> (cs*a - sn*b, sn*a + cs*b)
    const int sn = (dir == 1) - (dir == 3);
    const int cs = (dir == 0) - (dir == 2);

    // 9 rotated tap element-offsets ((a,b) are compile-time constants)
    int voff[9];
#pragma unroll
    for (int j = 0; j < 9; ++j) {
        const int a  = j / 3 - 1, b = j % 3 - 1;
        const int yy = (py + cs * a - sn * b) & W_MASK;
        const int xx = (px + sn * a + cs * b) & W_MASK;
        voff[j] = (yy << 10) | xx;
    }

    v16h a0, a1;
    if (hi == 0) gather_frags<0>(se, te, sr, hc, voff, a0, a1);
    else         gather_frags<8>(se, te, sr, hc, voff, a0, a1);

    // ---- Layer-1 weight B fragments from precomputed workspace (b128 loads).
    const v16h* wf = (const v16h*)wsf;
    const v16h bw00 = wf[0 * 32 + lane];
    const v16h bw01 = wf[1 * 32 + lane];
    const v16h bw10 = wf[2 * 32 + lane];
    const v16h bw11 = wf[3 * 32 + lane];
    const v16h b2f  = wf[4 * 32 + lane];

    // ---- Layer 1: (16x64) @ (64x32) via 4 WMMAs.
    v8f x0 = {}, x1 = {};
    x0 = __builtin_amdgcn_wmma_f32_16x16x32_f16(false, a0, false, bw00, (short)0, x0, false, false);
    x0 = __builtin_amdgcn_wmma_f32_16x16x32_f16(false, a1, false, bw01, (short)0, x0, false, false);
    x1 = __builtin_amdgcn_wmma_f32_16x16x32_f16(false, a0, false, bw10, (short)0, x1, false, false);
    x1 = __builtin_amdgcn_wmma_f32_16x16x32_f16(false, a1, false, bw11, (short)0, x1, false, false);

    // Bias + ReLU, stage activations to LDS as f16 (C/D layout: lane = N
    // feature, accumulator row r = pixel r + 8*hi).
    const float bb0 = b1[lm];
    const float bb1 = b1[lm + 16];
#pragma unroll
    for (int r = 0; r < 8; ++r) {
        const int m = r + 8 * hi;
        float v0 = x0[r] + bb0; v0 = v0 > 0.0f ? v0 : 0.0f;
        float v1 = x1[r] + bb1; v1 = v1 > 0.0f ? v1 : 0.0f;
        ldsX[wv][m][lm]      = (_Float16)v0;
        ldsX[wv][m][lm + 16] = (_Float16)v1;
    }
    __syncthreads();

    // ---- Layer 2: (16x32) @ (32x16) via 1 WMMA.
    // A2 fragment halves are two contiguous 8-half runs of row lm:
    // K = 8*hi + {0..7} and K = 8*hi + 16 + {0..7}  -> two ds_load_b128.
    const v8h* xrow = (const v8h*)(&ldsX[wv][lm][0]);
    const v8h alo = xrow[hi];
    const v8h ahi = xrow[hi + 2];
    v16h a2;
#pragma unroll
    for (int e = 0; e < 8; ++e) { a2[e] = alo[e]; a2[e + 8] = ahi[e]; }

    v8f y = {};
    y = __builtin_amdgcn_wmma_f32_16x16x32_f16(false, a2, false, b2f, (short)0, y, false, false);

    const float bb2 = (lm < 13) ? b2[lm] : 0.0f;
#pragma unroll
    for (int r = 0; r < 8; ++r) {
        ldsY[wv][r + 8 * hi][lm] = y[r] + bb2;
    }
    __syncthreads();

    // ---- Epilogue: one pixel per lane-half worker.
    // lanes 0..15: softmax over 9 logits + inverse-rotated scatter
    // lanes 16..31: tanh + 3 sigmoids for the same pixel set
    const int pp = p0 + lm;   // == p
    if (hi == 0) {
        float v9[9];
#pragma unroll
        for (int c = 0; c < 9; ++c) v9[c] = ldsY[wv][lm][c];
        float mx = v9[0];
#pragma unroll
        for (int c = 1; c < 9; ++c) mx = fmaxf(mx, v9[c]);
        float e9[9];
        float s = 0.0f;
#pragma unroll
        for (int c = 0; c < 9; ++c) { e9[c] = __expf(v9[c] - mx); s += e9[c]; }
        const float inv = 1.0f / s;
        // out[IROT[dir]^-1[i]] = props[i], and IROT[dir]^-1 = ROT[dir] which is
        // the same arithmetic rotation used for the gather.
#pragma unroll
        for (int i = 0; i < 9; ++i) {
            const int a  = i / 3 - 1, b = i % 3 - 1;
            const int dy = cs * a - sn * b;
            const int dx = sn * a + cs * b;
            const int jd = (dy + 1) * 3 + (dx + 1);
            out[jd * N_PIX + pp] = e9[i] * inv;
        }
    } else {
        out[9 * N_PIX + pp] = tanhf(ldsY[wv][lm][9]);
#pragma unroll
        for (int c = 0; c < 3; ++c) {
            out[(10 + c) * N_PIX + pp] =
                1.0f / (1.0f + __expf(-ldsY[wv][lm][10 + c]));
        }
    }
}

extern "C" void kernel_launch(void* const* d_in, const int* in_sizes, int n_in,
                              void* d_out, int out_size, void* d_ws, size_t ws_size,
                              hipStream_t stream) {
    const float* se   = (const float*)d_in[0];  // shareable_energy (W,W)
    const float* te   = (const float*)d_in[1];  // terrain          (W,W)
    const float* sr   = (const float*)d_in[2];  // sharing_rate     (W,W)
    const float* hc   = (const float*)d_in[3];  // hidden_channels  (3,W,W)
    const float* rotm = (const float*)d_in[4];  // rotation_matrix  (W,W)
    const float* w1   = (const float*)d_in[5];  // (32,6,3,3) == (32,54)
    const float* b1   = (const float*)d_in[6];  // (32)
    const float* w2   = (const float*)d_in[7];  // (13,32)
    const float* b2   = (const float*)d_in[8];  // (13)
    float* out        = (float*)d_out;          // 13*N floats

    _Float16* wsf = (_Float16*)d_ws;            // 5120 bytes used

    build_weight_frags<<<1, 32, 0, stream>>>(w1, w2, wsf);

    // 8 waves/block, 16 pixels/wave -> 128 pixels/block.
    dim3 block(256);
    dim3 grid(N_PIX / 128);
    energy_cnn_wmma_kernel<<<grid, block, 0, stream>>>(
        se, te, sr, hc, rotm, b1, b2, wsf, out);
}
